// VectorQuantizer_57097295233363
// MI455X (gfx1250) — matrix operand order repaired and checked by reading, not compiled, not run on hardware
//
#include <hip/hip_runtime.h>
#include <cstdint>
#include <cstddef>

// ---------------------------------------------------------------------------
// Vector-Quantizer for MI455X (gfx1250, wave32, WMMA + Tensor Data Mover).
//   prep kernel:  codebook f32 -> bf16 hi/lo split + ||w||^2, once, into d_ws
//   main kernel:  TDM-copies the 260KB prepared image into LDS (TENSORcnt),
//                 ranks ||w||^2 - 2*x.w with 3x bf16 WMMA per k-chunk
//                 (hi*hi + hi*lo + lo*hi ~ fp32-accurate argmin).
// ---------------------------------------------------------------------------

typedef __bf16 bf16;
typedef unsigned int u32;
typedef __attribute__((ext_vector_type(16))) __bf16 v16bf;
typedef __attribute__((ext_vector_type(8)))  float  v8f;
typedef __attribute__((ext_vector_type(4)))  u32    v4u;
typedef __attribute__((ext_vector_type(4)))  int    v4i;
typedef __attribute__((ext_vector_type(8)))  int    v8i;

#define NUM_CODES       1024
#define DIM             64
#define ROWS_PER_BLOCK  128     // 8 waves x 16 rows
#define NTHREADS        256

// LDS image == d_ws prepared image: [whi bf16 128K | wlo bf16 128K | wsq f32 4K]
#define WHI_OFF   0
#define WLO_OFF   (WHI_OFF + NUM_CODES * DIM * 2)        // 131072
#define WSQ_OFF   (WLO_OFF + NUM_CODES * DIM * 2)        // 262144
#define WIMG_BYTES (WSQ_OFF + NUM_CODES * 4)             // 266240 (= 16640 dw x 4 rows)
#define XHI_OFF   WIMG_BYTES                              // 266240
#define XLO_OFF   (XHI_OFF + ROWS_PER_BLOCK * DIM * 2)   // 282624
#define WIDX_OFF  (XLO_OFF + ROWS_PER_BLOCK * DIM * 2)   // 299008
#define SMEM_BYTES (WIDX_OFF + ROWS_PER_BLOCK * 4)       // 299520

// workspace layout
#define WS_PART_OFF  WIMG_BYTES                           // partial sums after image

#define TDM_ROW_DW   16640                                // dwords per D# row
#define TDM_NROWS    4

union FragAB { uint4 q[2]; v16bf v; };

__device__ __forceinline__ unsigned pack2(bf16 a, bf16 b) {
  union { bf16 h[2]; unsigned u; } c;
  c.h[0] = a; c.h[1] = b;
  return c.u;
}

// ---------------------------------------------------------------------------
// Prep: convert codebook once. Block b owns codes 16b..16b+15 (256 float4).
// ---------------------------------------------------------------------------
__global__ void __launch_bounds__(NTHREADS)
vq_prep_kernel(const float* __restrict__ W, char* __restrict__ ws)
{
  bf16*  whi = (bf16*)(ws + WHI_OFF);
  bf16*  wlo = (bf16*)(ws + WLO_OFF);
  float* wsq = (float*)(ws + WSQ_OFF);

  const int e4 = blockIdx.x * NTHREADS + threadIdx.x;   // 16384 float4 total
  const float4 wf = ((const float4*)W)[e4];
  const bf16 h0 = (bf16)wf.x, h1 = (bf16)wf.y, h2 = (bf16)wf.z, h3 = (bf16)wf.w;
  const bf16 l0 = (bf16)(wf.x - (float)h0);
  const bf16 l1 = (bf16)(wf.y - (float)h1);
  const bf16 l2 = (bf16)(wf.z - (float)h2);
  const bf16 l3 = (bf16)(wf.w - (float)h3);
  unsigned* dh = (unsigned*)(whi + (size_t)e4 * 4);
  dh[0] = pack2(h0, h1); dh[1] = pack2(h2, h3);
  unsigned* dl = (unsigned*)(wlo + (size_t)e4 * 4);
  dl[0] = pack2(l0, l1); dl[1] = pack2(l2, l3);

  if (threadIdx.x < 16) {
    const int row = blockIdx.x * 16 + threadIdx.x;
    float s = 0.f;
    for (int d = 0; d < DIM; ++d) { const float v = W[row * DIM + d]; s += v * v; }
    wsq[row] = s;
  }
}

// ---------------------------------------------------------------------------
// TDM: one DMA of the 260KB prepared image d_ws -> LDS offset 0.
// D#: data_size=4B, 2D tile 16640 dwords x 4 rows, stride 16640.
// ---------------------------------------------------------------------------
__device__ __forceinline__ void tdm_load_image_to_lds(u32 lds_addr, const void* gsrc)
{
#if __has_builtin(__builtin_amdgcn_tensor_load_to_lds)
  const unsigned long long ga = (unsigned long long)gsrc;
  v4u g0;
  g0.x = 1u;                                        // count=1, is_restore=0, no gather
  g0.y = lds_addr;                                  // lds_addr (bytes)
  g0.z = (u32)(ga & 0xffffffffull);                 // global_addr[31:0]
  g0.w = (u32)((ga >> 32) & 0x01ffffffull)          // global_addr[56:32]
       | (2u << 30);                                // type=2 ("image")
  v8i g1;
  g1[0] = (int)(2u << 16);                          // wg_mask=0, data_size=2 (4B)
  g1[1] = (int)((TDM_ROW_DW & 0xffffu) << 16);      // tensor_dim0[15:0]
  g1[2] = (int)((TDM_ROW_DW >> 16) & 0xffffu)       // tensor_dim0[31:16]
        | (int)((TDM_NROWS & 0xffffu) << 16);       // tensor_dim1[15:0]
  g1[3] = (int)((TDM_ROW_DW & 0xffffu) << 16);      // tensor_dim1[31:16]=0 | tile_dim0
  g1[4] = (int)(TDM_NROWS & 0xffffu);               // tile_dim1 | tile_dim2=0
  g1[5] = (int)TDM_ROW_DW;                          // tensor_dim0_stride[31:0]
  g1[6] = 0;                                        // stride[47:32] | dim1_stride lo
  g1[7] = 0;
  const v4i z4 = {0, 0, 0, 0};
#if __clang_major__ >= 23
  const v8i z8 = {0, 0, 0, 0, 0, 0, 0, 0};
  __builtin_amdgcn_tensor_load_to_lds(g0, g1, z4, z4, z8, 0);
#else
  __builtin_amdgcn_tensor_load_to_lds(g0, g1, z4, z4, 0);
#endif
  __builtin_amdgcn_s_wait_tensorcnt(0);
#else
  (void)lds_addr; (void)gsrc;
#endif
}

__global__ void __launch_bounds__(NTHREADS)
vq_main_kernel(const float* __restrict__ X, const float* __restrict__ W,
               const char* __restrict__ wimg,
               float* __restrict__ outQ, float* __restrict__ outIdx,
               float* __restrict__ partials)
{
  extern __shared__ __align__(32) char smem[];
  bf16*  whi  = (bf16*)(smem + WHI_OFF);
  bf16*  wlo  = (bf16*)(smem + WLO_OFF);
  float* wsq  = (float*)(smem + WSQ_OFF);
  bf16*  xhi  = (bf16*)(smem + XHI_OFF);
  bf16*  xlo  = (bf16*)(smem + XLO_OFF);
  int*   widx = (int*)(smem + WIDX_OFF);
  float* red  = (float*)(smem + WSQ_OFF);   // reused for loss reduction

  const int tid   = threadIdx.x;
  const int lane  = tid & 31;
  const int wv    = tid >> 5;
  const int m     = lane & 15;
  const int half  = lane >> 4;
  const int grow0 = blockIdx.x * ROWS_PER_BLOCK;

  // ---- DMA the prepared codebook image into LDS (wave 0 issues, all wait) ----
#if __has_builtin(__builtin_amdgcn_tensor_load_to_lds)
  if (wv == 0) {
    tdm_load_image_to_lds((u32)(size_t)smem, wimg);
  }
#else
  for (int i = 0; i < (int)(WIMG_BYTES / 16) / NTHREADS; ++i) {
    const int e = tid + NTHREADS * i;
    ((uint4*)smem)[e] = ((const uint4*)wimg)[e];
  }
  { const int e = tid + NTHREADS * ((int)(WIMG_BYTES / 16) / NTHREADS);
    if (e < (int)(WIMG_BYTES / 16)) ((uint4*)smem)[e] = ((const uint4*)wimg)[e]; }
#endif

  // ---- Stage this block's 128x64 x-tile as bf16 hi/lo split ----
  const float* Xblk = X + (size_t)grow0 * DIM;
  for (int i = 0; i < 8; ++i) {
    const int e4  = tid + NTHREADS * i;   // 2048 float4 total
    const float4 xf = ((const float4*)Xblk)[e4];
    const bf16 h0 = (bf16)xf.x, h1 = (bf16)xf.y, h2 = (bf16)xf.z, h3 = (bf16)xf.w;
    const bf16 l0 = (bf16)(xf.x - (float)h0);
    const bf16 l1 = (bf16)(xf.y - (float)h1);
    const bf16 l2 = (bf16)(xf.z - (float)h2);
    const bf16 l3 = (bf16)(xf.w - (float)h3);
    unsigned* dh = (unsigned*)(xhi + (size_t)e4 * 4);
    dh[0] = pack2(h0, h1); dh[1] = pack2(h2, h3);
    unsigned* dl = (unsigned*)(xlo + (size_t)e4 * 4);
    dl[0] = pack2(l0, l1); dl[1] = pack2(l2, l3);
  }
  __syncthreads();

  // ---- This wave's A fragments (16 rows x 64 dims, hi & lo) ----
  // A 16x32 bf16: lanes 0-15 M=lane, K={0..7,16..23}; lanes 16-31 M=lane-16, K={8..15,24..31}
  const int lr = wv * 16 + m;
  FragAB ahi[2], alo[2];
#pragma unroll
  for (int c = 0; c < 2; ++c) {
    const int s1 = 32 * c + half * 8;
    ahi[c].q[0] = *(const uint4*)(xhi + lr * DIM + s1);
    ahi[c].q[1] = *(const uint4*)(xhi + lr * DIM + s1 + 16);
    alo[c].q[0] = *(const uint4*)(xlo + lr * DIM + s1);
    alo[c].q[1] = *(const uint4*)(xlo + lr * DIM + s1 + 16);
  }

  // ---- Sweep 64 code tiles: 6 WMMAs each on two independent accumulators ----
  float best[8];
  int   bidx[8];
#pragma unroll
  for (int j = 0; j < 8; ++j) { best[j] = 3.4e38f; bidx[j] = 0; }

#pragma unroll 2
  for (int t = 0; t < NUM_CODES / 16; ++t) {
    const int n = t * 16 + m;              // this lane's code column (global)
    v8f acc0 = {0.f, 0.f, 0.f, 0.f, 0.f, 0.f, 0.f, 0.f};   // hi*hi
    v8f acc1 = {0.f, 0.f, 0.f, 0.f, 0.f, 0.f, 0.f, 0.f};   // corrections
#pragma unroll
    for (int c = 0; c < 2; ++c) {
      // B 32x16 bf16: lanes 0-15 N=lane, K={0..7,16..23}; lanes 16-31 N=lane-16, K={8..15,24..31}
      const int kb = 32 * c + half * 8;
      FragAB bhi, blo;
      bhi.q[0] = *(const uint4*)(whi + n * DIM + kb);
      bhi.q[1] = *(const uint4*)(whi + n * DIM + kb + 16);
      blo.q[0] = *(const uint4*)(wlo + n * DIM + kb);
      blo.q[1] = *(const uint4*)(wlo + n * DIM + kb + 16);
      acc0 = __builtin_amdgcn_wmma_f32_16x16x32_bf16(false, ahi[c].v, false, bhi.v,
                                                     (short)0, acc0, false, false);
      acc1 = __builtin_amdgcn_wmma_f32_16x16x32_bf16(false, ahi[c].v, false, blo.v,
                                                     (short)0, acc1, false, false);
      acc1 = __builtin_amdgcn_wmma_f32_16x16x32_bf16(false, alo[c].v, false, bhi.v,
                                                     (short)0, acc1, false, false);
    }
    const float wsv = wsq[n];
#pragma unroll
    for (int j = 0; j < 8; ++j) {
      const float dj = wsv - 2.0f * (acc0[j] + acc1[j]);  // ||x||^2 constant per row
      if (dj < best[j]) { best[j] = dj; bidx[j] = n; }    // ascending n -> first-index ties
    }
  }

  // ---- Cross-lane argmin over the 16 N-residue lanes (first index on ties) ----
#pragma unroll
  for (int j = 0; j < 8; ++j) {
    float d = best[j];
    int   i = bidx[j];
#pragma unroll
    for (int mask = 1; mask < 16; mask <<= 1) {
      const float od = __shfl_xor(d, mask, 32);
      const int   oi = __shfl_xor(i, mask, 32);
      if (od < d || (od == d && oi < i)) { d = od; i = oi; }
    }
    if (m == 0) {
      const int rowm = j + half * 8;   // C/D layout: VGPR j -> row j (+8 upper half)
      widx[wv * 16 + rowm] = i;
      outIdx[grow0 + wv * 16 + rowm] = (float)i;
    }
  }
  __syncthreads();

  // ---- Gather quantized rows (exact f32 codebook) + loss partial ----
  float lsum = 0.f;
  for (int i = 0; i < 32; ++i) {
    const int e   = tid + NTHREADS * i;    // 8192 elements
    const int row = e >> 6;
    const int col = e & 63;
    const int ci  = widx[row];
    const float qv = W[(size_t)ci * DIM + col];
    const float xv = X[(size_t)(grow0 + row) * DIM + col];
    outQ[(size_t)(grow0 + row) * DIM + col] = qv;
    const float df = qv - xv;
    lsum += df * df;
  }
  red[tid] = lsum;
  __syncthreads();
  for (int off = NTHREADS / 2; off > 0; off >>= 1) {
    if (tid < off) red[tid] += red[tid + off];
    __syncthreads();
  }
  if (tid == 0) partials[blockIdx.x] = red[0];
}

__global__ void __launch_bounds__(256)
vq_finalize_kernel(const float* __restrict__ partials, int nblocks,
                   float inv_cnt, float* __restrict__ outLoss)
{
  __shared__ float red[256];
  const int t = threadIdx.x;
  float s = 0.f;
  for (int i = t; i < nblocks; i += 256) s += partials[i];
  red[t] = s;
  __syncthreads();
  for (int off = 128; off > 0; off >>= 1) {
    if (t < off) red[t] += red[t + off];
    __syncthreads();
  }
  // loss = q_latent + 0.25*e_latent, both numerically mean((q-x)^2)
  if (t == 0) outLoss[0] = 1.25f * red[0] * inv_cnt;
}

extern "C" void kernel_launch(void* const* d_in, const int* in_sizes, int n_in,
                              void* d_out, int out_size, void* d_ws, size_t ws_size,
                              hipStream_t stream) {
  (void)n_in; (void)out_size; (void)ws_size;
  const float* X = (const float*)d_in[0];     // [16,4096,64] f32
  const float* W = (const float*)d_in[1];     // [1024,64]    f32

  const int N      = in_sizes[0] / DIM;             // 65536 rows
  const int blocks = N / ROWS_PER_BLOCK;            // 512

  float* out      = (float*)d_out;
  float* outQ     = out;                            // N*DIM
  float* outIdx   = out + (size_t)N * DIM;          // N
  float* outLoss  = outIdx + N;                     // 1
  char*  ws       = (char*)d_ws;
  float* partials = (float*)(ws + WS_PART_OFF);     // blocks floats

  vq_prep_kernel<<<NUM_CODES * DIM / 4 / NTHREADS, NTHREADS, 0, stream>>>(W, ws);
  vq_main_kernel<<<blocks, NTHREADS, SMEM_BYTES, stream>>>(X, W, ws, outQ, outIdx, partials);
  vq_finalize_kernel<<<1, 256, 0, stream>>>(partials, blocks,
                                            1.0f / (float)((size_t)N * DIM), outLoss);
}
